// LDABBVI_27127013441903
// MI455X (gfx1250) — matrix operand order repaired and checked by reading, not compiled, not run on hardware
//
#include <hip/hip_runtime.h>
#include <cstdint>

#define VOCABSZ 50000
#define KTOP    64
#define DDOC    256
#define SSAMP   4
#define EPSLL   1e-10f
#define VT_PER_BLK 8   // v-tiles (16 cols each) per block in the GEMM

typedef __attribute__((ext_vector_type(16))) __bf16 v16bf;
typedef __attribute__((ext_vector_type(8)))  float  v8f;
typedef __attribute__((ext_vector_type(4)))  unsigned int v4u;
typedef __attribute__((ext_vector_type(8)))  int   v8i_;
typedef __attribute__((ext_vector_type(4)))  int   v4i_;

// ---------------- RNG / math helpers ----------------
__device__ __forceinline__ uint32_t hash32(uint32_t x) {
    x ^= x >> 16; x *= 0x7feb352du;
    x ^= x >> 15; x *= 0x846ca68bu;
    x ^= x >> 16; return x;
}
__device__ __forceinline__ float next_u(uint32_t& st) {
    st = st * 1664525u + 1013904223u;
    uint32_t h = hash32(st);
    return (float)(h >> 8) * 5.9604645e-08f + 2.9802322e-08f; // (0,1)
}
__device__ __forceinline__ uint16_t f2bf(float f) {
    uint32_t b = __float_as_uint(f);
    uint32_t r = b + 0x7fffu + ((b >> 16) & 1u); // RNE
    return (uint16_t)(r >> 16);
}

// Marsaglia-Tsang gamma sampler with small-shape boost done in log space.
__device__ float gamma_sample(float a, uint32_t sd, uint32_t idx, float* lg_out) {
    uint32_t st = hash32(sd ^ hash32(idx * 0x9e3779b9u + 0x85ebca6bu));
    float boost = 0.f, ap = a;
    if (a < 1.f) { ap = a + 1.f; boost = __logf(next_u(st)) / a; }
    float d = ap - 0.33333333f;
    float c = 0.33333333f * rsqrtf(d);
    float g = d;
    #pragma unroll 1
    for (int it = 0; it < 16; ++it) {
        float u1 = next_u(st), u2 = next_u(st);
        float x = __fsqrt_rn(-2.f * __logf(u1)) * __cosf(6.28318531f * u2);
        float v = 1.f + c * x;
        if (v <= 0.f) continue;
        v = v * v * v;
        float u = next_u(st);
        if (__logf(u) < 0.5f * x * x + d - d * v + d * __logf(v)) { g = d * v; break; }
    }
    float lg = __logf(g) + boost;
    *lg_out = lg;
    return fmaxf(__expf(fmaxf(lg, -85.f)), 1e-37f);
}

// ---------------- Kernel 1: zero stat buffers ----------------
__global__ void k_zero(float* acc, float* lam_sum, float* lam_gl, float* gsum, float* glog) {
    int t = threadIdx.x;
    if (t < 8) acc[t] = 0.f;
    if (t < KTOP) { lam_sum[t] = 0.f; lam_gl[t] = 0.f; }
    if (t < SSAMP * KTOP) { gsum[t] = 0.f; glog[t] = 0.f; }
}

// ---------------- Kernel 2: per-k lambda stats ----------------
__global__ void k_lam_stats(const float* __restrict__ tlv, float* lam_sum, float* lam_gl) {
    __shared__ float r1[256], r2[256];
    int k = blockIdx.y;
    int base = blockIdx.x * 1024 + threadIdx.x;
    float s1 = 0.f, s2 = 0.f;
    #pragma unroll
    for (int i = 0; i < 4; ++i) {
        int v = base + i * 256;
        if (v < VOCABSZ) {
            float lam = __expf(tlv[(size_t)k * VOCABSZ + v]);
            s1 += lam; s2 += lgammaf(lam);
        }
    }
    r1[threadIdx.x] = s1; r2[threadIdx.x] = s2;
    for (int off = 128; off > 0; off >>= 1) {
        __syncthreads();
        if (threadIdx.x < off) { r1[threadIdx.x] += r1[threadIdx.x + off]; r2[threadIdx.x] += r2[threadIdx.x + off]; }
    }
    if (threadIdx.x == 0) { atomicAdd(&lam_sum[k], r1[0]); atomicAdd(&lam_gl[k], r2[0]); }
}

// ---------------- Kernel 3: doc-side sampling + normalize + log_q ----------------
__global__ void k_doc(const float* __restrict__ dlv, float* __restrict__ theta, float* acc) {
    __shared__ float r1[KTOP], r2[KTOP];
    int d = blockIdx.x, s = blockIdx.y, k = threadIdx.x;
    float gam = __expf(dlv[d * KTOP + k]);
    float lg;
    float g = gamma_sample(gam, 0x9000u + s, (uint32_t)(d * KTOP + k), &lg);
    r1[k] = g; r2[k] = gam;
    for (int off = 32; off > 0; off >>= 1) {
        __syncthreads();
        if (k < off) { r1[k] += r1[k + off]; r2[k] += r2[k + off]; }
    }
    __syncthreads();
    float Gs = r1[0], gamS = r2[0];
    __syncthreads();
    theta[((size_t)s * DDOC + d) * KTOP + k] = g / Gs;
    r1[k] = (gam - 1.f) * (lg - __logf(Gs)) - lgammaf(gam);
    for (int off = 32; off > 0; off >>= 1) {
        __syncthreads();
        if (k < off) r1[k] += r1[k + off];
    }
    __syncthreads();
    if (k == 0) atomicAdd(&acc[1], r1[0] + lgammaf(gamS)); // log_q doc part
}

// ---------------- Kernel 4: topic sampling, LDS transpose to [s][v][k] bf16 ----------------
__global__ void k_topic_sample(const float* __restrict__ tlv, uint16_t* __restrict__ tT,
                               float* gsum, float* glog) {
    __shared__ __align__(16) uint16_t tile[64][72];
    __shared__ float ksum[KTOP], klog[KTOP];
    int s = blockIdx.y;
    int v0 = blockIdx.x * 64;
    int t = threadIdx.x;
    if (t < KTOP) { ksum[t] = 0.f; klog[t] = 0.f; }
    __syncthreads();
    #pragma unroll 1
    for (int it = 0; it < 16; ++it) {
        int k = it * 4 + (t >> 6);
        int v = v0 + (t & 63);
        if (v < VOCABSZ) {
            float lam = __expf(tlv[(size_t)k * VOCABSZ + v]);
            float lg;
            float g = gamma_sample(lam, 0x2000u + s, (uint32_t)(k * VOCABSZ + v), &lg);
            tile[t & 63][k] = f2bf(g);
            atomicAdd(&ksum[k], g);
            atomicAdd(&klog[k], (lam - 1.f) * lg);
        }
    }
    __syncthreads();
    if (t < KTOP) {
        atomicAdd(&gsum[s * KTOP + t], ksum[t]);
        atomicAdd(&glog[s * KTOP + t], klog[t]);
    }
    int row = t >> 2, cc = t & 3;
    int v = v0 + row;
    if (v < VOCABSZ) {
        const uint4* src = (const uint4*)&tile[row][cc * 16];
        uint4* dst = (uint4*)(tT + ((size_t)s * VOCABSZ + v) * KTOP + cc * 16);
        dst[0] = src[0]; dst[1] = src[1];
    }
}

// ---------------- Kernel 5: per-(s,k) log_q finalize + inv sums ----------------
__global__ void k_topic_fin(const float* gsum, const float* glog, const float* lam_sum,
                            const float* lam_gl, float* invg, float* acc) {
    int i = threadIdx.x;
    int k = i & (KTOP - 1);
    float S = gsum[i];
    float lq = glog[i] - (lam_sum[k] - (float)VOCABSZ) * __logf(S)
             + lgammaf(lam_sum[k]) - lam_gl[k];
    atomicAdd(&acc[1], lq);
    invg[i] = 1.f / S;
}

// ---------------- Kernel 6: A = bf16(theta * 1/Sum_topic) ----------------
__global__ void k_scaleA(const float* __restrict__ theta, const float* __restrict__ invg,
                         uint16_t* __restrict__ Ab) {
    int idx = blockIdx.x * 256 + threadIdx.x;
    int k = idx & (KTOP - 1);
    int s = idx >> 14;
    Ab[idx] = f2bf(theta[idx] * invg[(s << 6) | k]);
}

// ---------------- TDM: async-load one 16x128B B-tile into LDS (padded rows) ----------------
// D# per ISA ch.8: 2D tile of 8-byte elements, 16 wide x 16 rows; pad 16B after each 128B
// so LDS rows land at stride 144B (bank-conflict-free 16-lane b128 reads).
__device__ __forceinline__ void tdm_load_B(const uint16_t* gptr, uint32_t lds_off) {
    uint64_t ga = (uint64_t)(uintptr_t)gptr;
    v4u g0;
    g0[0] = 1u;                                              // count=1, user descriptor
    g0[1] = lds_off;                                         // LDS byte address
    g0[2] = (uint32_t)(ga & 0xffffffffu);                    // global_addr[31:0]
    g0[3] = (uint32_t)((ga >> 32) & 0x1ffffffu) | (2u << 30); // addr[56:32] | type=2
    v8i_ g1;
    g1[0] = (int)((3u << 16) | (1u << 20) | (4u << 22) | (3u << 25));
    //            data_size=8B | pad_en   | intv=32dw   | amount=4dw(16B)
    g1[1] = (int)(16u << 16);   // tensor_dim0 = 16 (8B elems per row)
    g1[2] = (int)(16u << 16);   // tensor_dim1 = 16 rows
    g1[3] = (int)(16u << 16);   // tile_dim0 = 16
    g1[4] = 16;                 // tile_dim1 = 16, tile_dim2 = 0
    g1[5] = 16;                 // tensor_dim0_stride = 16
    g1[6] = 0;
    g1[7] = 0;
    v4i_ z4 = {0, 0, 0, 0};
    v8i_ z8 = {0, 0, 0, 0, 0, 0, 0, 0};
    // 6-arg form (clang-23 / therock toolchain): (g0, g1, g2, g3, g4, cpol)
    __builtin_amdgcn_tensor_load_to_lds(g0, g1, z4, z4, z8, 0);
}

// ---------------- Kernel 7: fused WMMA GEMM + masked log-likelihood ----------------
// Block: 8 waves, handles VT_PER_BLK v-tiles; wave w owns d-tiles {2w, 2w+1} (A regs
// hoisted out of the v-loop). B tiles stream HBM->LDS via TDM, double-buffered.
__global__ void __launch_bounds__(256)
k_gemm_ll(const float* __restrict__ docs, const uint16_t* __restrict__ tT,
          const uint16_t* __restrict__ Ab, float* acc) {
    union U { v16bf v; uint4 q[2]; };
    __shared__ __align__(16) uint16_t Bb[2][16 * 72];   // 16 rows @ 144B stride
    __shared__ float wsum[8];
    int s = blockIdx.y;
    int vbase = blockIdx.x * (16 * VT_PER_BLK);
    int nt = (VOCABSZ - vbase) / 16; if (nt > VT_PER_BLK) nt = VT_PER_BLK;
    int w = threadIdx.x >> 5, lane = threadIdx.x & 31;

    // ---- hoisted A operands (per-wave d-tiles, constant over v) ----
    U alo[2], ahi[2];
    int ko = (lane < 16) ? 0 : 8;
    #pragma unroll
    for (int tile = 0; tile < 2; ++tile) {
        int dr = (w * 2 + tile) * 16 + (lane & 15);
        const uint16_t* ap = Ab + (((size_t)s * DDOC + dr) * KTOP + ko);
        alo[tile].q[0] = *(const uint4*)ap;        alo[tile].q[1] = *(const uint4*)(ap + 16);
        ahi[tile].q[0] = *(const uint4*)(ap + 32); ahi[tile].q[1] = *(const uint4*)(ap + 48);
    }

    // ---- TDM prefetch of first B tile ----
    if (threadIdx.x < 32) {
        tdm_load_B(tT + ((size_t)s * VOCABSZ + vbase) * KTOP,
                   (uint32_t)(uintptr_t)&Bb[0][0]);
    }

    int kb = (lane < 16) ? 0 : 8;
    float lacc = 0.f;
    #pragma unroll 1
    for (int t = 0; t < nt; ++t) {
        if (threadIdx.x < 32) {
            if (t + 1 < nt) {
                tdm_load_B(tT + ((size_t)s * VOCABSZ + vbase + (t + 1) * 16) * KTOP,
                           (uint32_t)(uintptr_t)&Bb[(t + 1) & 1][0]);
                __builtin_amdgcn_s_wait_tensorcnt(1);   // tile t landed
            } else {
                __builtin_amdgcn_s_wait_tensorcnt(0);
            }
        }
        __syncthreads();                                // B tile t visible to all waves

        // B operand from LDS (rows at 144B stride -> conflict-free)
        const uint16_t* bp = &Bb[t & 1][(lane & 15) * 72 + kb];
        U blo, bhi;
        blo.q[0] = *(const uint4*)bp;        blo.q[1] = *(const uint4*)(bp + 16);
        bhi.q[0] = *(const uint4*)(bp + 32); bhi.q[1] = *(const uint4*)(bp + 48);

        int v0 = vbase + t * 16;
        int vv = v0 + (lane & 15);
        #pragma unroll
        for (int tile = 0; tile < 2; ++tile) {
            v8f c = {};
            c = __builtin_amdgcn_wmma_f32_16x16x32_bf16(false, alo[tile].v, false, blo.v,
                                                        (short)0, c, false, false);
            c = __builtin_amdgcn_wmma_f32_16x16x32_bf16(false, ahi[tile].v, false, bhi.v,
                                                        (short)0, c, false, false);
            int dbase = (w * 2 + tile) * 16 + ((lane < 16) ? 0 : 8);
            #pragma unroll
            for (int r = 0; r < 8; ++r) {
                float dc = docs[(size_t)(dbase + r) * VOCABSZ + vv];
                float lp = __logf(c[r] + EPSLL);         // branchless epilogue
                lacc += (dc > 0.f ? dc : 0.f) * lp;
            }
        }
        __syncthreads();                                // done reading buffer t&1
    }

    for (int m = 16; m > 0; m >>= 1) lacc += __shfl_xor(lacc, m, 32);
    if (lane == 0) wsum[w] = lacc;
    __syncthreads();
    if (threadIdx.x == 0) {
        float tt = 0.f;
        #pragma unroll
        for (int i = 0; i < 8; ++i) tt += wsum[i];
        atomicAdd(&acc[0], tt);
    }
}

// ---------------- Kernel 8: combine (analytic priors + sampled terms) ----------------
__global__ void k_final(const float* acc, const int* nptr, float* out) {
    float S = (float)nptr[0];
    float elbo = (float)KTOP * lgammaf((float)VOCABSZ)
               + (float)DDOC * lgammaf((float)KTOP)
               + (acc[0] - acc[1]) / S;
    out[0] = -elbo;
}

// ---------------- Host launcher ----------------
extern "C" void kernel_launch(void* const* d_in, const int* in_sizes, int n_in,
                              void* d_out, int out_size, void* d_ws, size_t ws_size,
                              hipStream_t stream) {
    const float* docs = (const float*)d_in[0];
    const float* tlv  = (const float*)d_in[1];
    const float* dlv  = (const float*)d_in[2];
    const int*   nptr = (const int*)d_in[3];
    float* out = (float*)d_out;

    char* w = (char*)d_ws;
    size_t o = 0;
    uint16_t* tT = (uint16_t*)(w + o);  o += (size_t)SSAMP * VOCABSZ * KTOP * 2; // 25.6 MB
    float* theta = (float*)(w + o);     o += (size_t)SSAMP * DDOC * KTOP * 4;
    uint16_t* Ab = (uint16_t*)(w + o);  o += (size_t)SSAMP * DDOC * KTOP * 2;
    float* lam_sum = (float*)(w + o);   o += 256;
    float* lam_gl  = (float*)(w + o);   o += 256;
    float* gsum    = (float*)(w + o);   o += 1024;
    float* glog    = (float*)(w + o);   o += 1024;
    float* invg    = (float*)(w + o);   o += 1024;
    float* acc     = (float*)(w + o);   o += 256;

    k_zero<<<1, 1024, 0, stream>>>(acc, lam_sum, lam_gl, gsum, glog);
    k_lam_stats<<<dim3((VOCABSZ + 1023) / 1024, KTOP), 256, 0, stream>>>(tlv, lam_sum, lam_gl);
    k_doc<<<dim3(DDOC, SSAMP), KTOP, 0, stream>>>(dlv, theta, acc);
    k_topic_sample<<<dim3((VOCABSZ + 63) / 64, SSAMP), 256, 0, stream>>>(tlv, tT, gsum, glog);
    k_topic_fin<<<1, SSAMP * KTOP, 0, stream>>>(gsum, glog, lam_sum, lam_gl, invg, acc);
    k_scaleA<<<(SSAMP * DDOC * KTOP) / 256, 256, 0, stream>>>(theta, invg, Ab);
    k_gemm_ll<<<dim3((VOCABSZ + 16 * VT_PER_BLK - 1) / (16 * VT_PER_BLK), SSAMP),
                256, 0, stream>>>(docs, tT, Ab, acc);
    k_final<<<1, 1, 0, stream>>>(acc, nptr, out);
}
